// SelectiveLinear_28303834480975
// MI455X (gfx1250) — matrix operand order repaired and checked by reading, not compiled, not run on hardware
//
#include <hip/hip_runtime.h>
#include <hip/hip_bf16.h>

typedef __attribute__((ext_vector_type(16))) _Float16 v16h;
typedef __attribute__((ext_vector_type(8)))  _Float16 v8h;
typedef __attribute__((ext_vector_type(4)))  _Float16 v4h;
typedef __attribute__((ext_vector_type(8)))  float    v8f;

#define DIN    1024
#define DOUT   1024
#define NEXP   8
#define TOKENS 8192          // B*S = 4*2048
#define TILE_M 256
#define TILE_N 128
#define TILE_K 32
#define LDA    40            // padded halves per LDS row (20 dwords -> conflict-free)
#define LDB    40

// Branch-free tanh: 1 - 2*rcp(exp(2x)+1).
// v_mul + v_exp + v_add + v_rcp + v_fma; saturates correctly at +/-inf,
// absolute error ~1e-7 near 0 (well under the f16-GEMM noise floor).
__device__ __forceinline__ float fast_tanh(float v) {
    float e = __expf(2.0f * v);
    return 1.0f - 2.0f * __builtin_amdgcn_rcpf(e + 1.0f);
}

__global__ __launch_bounds__(256, 1)
void selective_linear_kernel(const float* __restrict__ x,
                             const float* __restrict__ sel,
                             const float* __restrict__ w,
                             const float* __restrict__ bias,
                             float* __restrict__ out)
{
    __shared__ __align__(16) _Float16 As[2][TILE_M * LDA];   // 40 KB
    __shared__ __align__(16) _Float16 Bs[2][TILE_N * LDB];   // 20 KB

    const int tid   = threadIdx.x;
    const int lane  = tid & 31;
    const int wv    = tid >> 5;
    const int wm    = (wv & 3) * 64;   // wave tile: 64 tokens
    const int wn    = (wv >> 2) * 64;  //            64 dout
    const int mbase = blockIdx.x * TILE_M;
    const int nbase = blockIdx.y * TILE_N;
    const int b     = mbase >> 11;     // token / 2048 -> batch (tile never crosses batch)

    // softmax normalization constants over the 8 logits of batch b
    float mx = -3.0e38f;
    #pragma unroll
    for (int n = 0; n < NEXP; ++n) mx = fmaxf(mx, sel[b * NEXP + n]);
    float ssum = 0.0f;
    #pragma unroll
    for (int n = 0; n < NEXP; ++n) ssum += __expf(sel[b * NEXP + n] - mx);
    const float inv = 1.0f / ssum;

    // Load one (expert n, k-chunk) pair of tiles, converting f32 -> f16 into LDS buffer bufid.
    auto load_tiles = [&](int n, int kc, int bufid) {
        const float* xs = x + (size_t)mbase * DIN + kc * TILE_K;
        #pragma unroll
        for (int i = 0; i < 8; ++i) {                 // 2048 float4 / 256 threads
            int f = i * 256 + tid;
            int row = f >> 3, cv = f & 7;
            float4 v = *(const float4*)(xs + row * DIN + cv * 4);
            v4h h = { (_Float16)v.x, (_Float16)v.y, (_Float16)v.z, (_Float16)v.w };
            *(v4h*)(&As[bufid][row * LDA + cv * 4]) = h;
        }
        const float* wsrc = w + (size_t)n * (DOUT * DIN) + (size_t)nbase * DIN + kc * TILE_K;
        #pragma unroll
        for (int i = 0; i < 4; ++i) {                 // 1024 float4 / 256 threads
            int f = i * 256 + tid;
            int row = f >> 3, cv = f & 7;
            float4 v = *(const float4*)(wsrc + row * DIN + cv * 4);
            v4h h = { (_Float16)v.x, (_Float16)v.y, (_Float16)v.z, (_Float16)v.w };
            *(v4h*)(&Bs[bufid][row * LDB + cv * 4]) = h;
        }
    };

    v8f c[4][4] = {};   // per-expert GEMM accumulators (f32)
    v8f o[4][4] = {};   // softmax-weighted tanh accumulators (f32)
    const int  lrow = lane & 15;
    const bool lo16 = lane < 16;

    load_tiles(0, 0, 0);
    __syncthreads();

    const int NITER = NEXP * (DIN / TILE_K);  // 8 experts * 32 k-chunks = 256
    for (int it = 0; it < NITER; ++it) {
        const int buf = it & 1;
        if (it + 1 < NITER)
            load_tiles((it + 1) >> 5, (it + 1) & 31, buf ^ 1);

        // A fragments: per lane, row M, K in {0-7,16-23} (lanes 0-15) / {8-15,24-31} (lanes 16-31)
        v16h Af[4], Bf[4];
        #pragma unroll
        for (int i = 0; i < 4; ++i) {
            const _Float16* p = &As[buf][(wm + i * 16 + lrow) * LDA + (lo16 ? 0 : 8)];
            v8h lo = *(const v8h*)p;
            v8h hi = *(const v8h*)(p + 16);
            Af[i] = __builtin_shufflevector(lo, hi, 0,1,2,3,4,5,6,7,8,9,10,11,12,13,14,15);
        }
        // B fragments: per lane, column N, K in {0-7,16-23} (lanes 0-15) / {8-15,24-31} (lanes 16-31)
        #pragma unroll
        for (int j = 0; j < 4; ++j) {
            const _Float16* p = &Bs[buf][(wn + j * 16 + lrow) * LDB + (lo16 ? 0 : 8)];
            v8h lo = *(const v8h*)p;
            v8h hi = *(const v8h*)(p + 16);
            Bf[j] = __builtin_shufflevector(lo, hi, 0,1,2,3,4,5,6,7,8,9,10,11,12,13,14,15);
        }

        #pragma unroll
        for (int i = 0; i < 4; ++i)
            #pragma unroll
            for (int j = 0; j < 4; ++j)
                c[i][j] = __builtin_amdgcn_wmma_f32_16x16x32_f16(
                    false, Af[i], false, Bf[j], (short)0, c[i][j], false, false);

        if ((it & 31) == 31) {  // expert finished: o += p_n * tanh(c); c = 0
            const int n = it >> 5;
            const float pn = __expf(sel[b * NEXP + n] - mx) * inv;
            #pragma unroll
            for (int i = 0; i < 4; ++i)
                #pragma unroll
                for (int j = 0; j < 4; ++j)
                    #pragma unroll
                    for (int r = 0; r < 8; ++r) {
                        o[i][j][r] += pn * fast_tanh(c[i][j][r]);
                        c[i][j][r] = 0.0f;
                    }
        }
        __syncthreads();
    }

    // Epilogue: add tanh(sum_n p_n * bias[n, col]) and store f32 output.
    #pragma unroll
    for (int j = 0; j < 4; ++j) {
        const int col = nbase + wn + j * 16 + lrow;
        float acc = 0.0f;
        #pragma unroll
        for (int n = 0; n < NEXP; ++n)
            acc += __expf(sel[b * NEXP + n] - mx) * inv * bias[n * DOUT + col];
        const float wb = fast_tanh(acc);
        #pragma unroll
        for (int i = 0; i < 4; ++i) {
            const int row0 = mbase + wm + i * 16 + (lo16 ? 0 : 8);
            #pragma unroll
            for (int r = 0; r < 8; ++r)
                out[(size_t)(row0 + r) * DOUT + col] = o[i][j][r] + wb;
        }
    }
}

extern "C" void kernel_launch(void* const* d_in, const int* in_sizes, int n_in,
                              void* d_out, int out_size, void* d_ws, size_t ws_size,
                              hipStream_t stream) {
    (void)in_sizes; (void)n_in; (void)out_size; (void)d_ws; (void)ws_size;
    const float* x    = (const float*)d_in[0];
    const float* sel  = (const float*)d_in[1];
    const float* w    = (const float*)d_in[2];
    const float* bias = (const float*)d_in[3];
    float* out        = (float*)d_out;

    dim3 grid(TOKENS / TILE_M, DOUT / TILE_N);  // 32 x 8 = 256 workgroups
    selective_linear_kernel<<<grid, 256, 0, stream>>>(x, sel, w, bias, out);
}
